// GroupedExperts_52707838657311
// MI455X (gfx1250) — matrix operand order repaired and checked by reading, not compiled, not run on hardware
//
#include <hip/hip_runtime.h>
#include <hip/hip_bf16.h>
#include <stdint.h>

// ---------------- problem constants (from reference) ----------------
#define N_TOK    4096
#define DIM      2048
#define NEXP     32
#define TOPK     4
#define INTER    1408
#define TWO_INTER (2*INTER)          // 2816
#define T_ASSIGN (N_TOK*TOPK)        // 16384
#define CAP      1024                // 2 * ceil(T_ASSIGN/NEXP)

#define ALPHA_C      1.702f
#define LIMIT_C      7.0f
#define LIN_OFFSET_C 1.0f

typedef __bf16 bf16;
typedef __attribute__((ext_vector_type(2)))  __bf16 bf16x2;
typedef __attribute__((ext_vector_type(16))) __bf16 v16bf;
typedef __attribute__((ext_vector_type(8)))  float  v8f;
typedef __attribute__((ext_vector_type(4)))  float  vf4;
typedef __attribute__((ext_vector_type(2)))  unsigned int vu2;

union FragU { v16bf v; uint4 q[2]; };

// pack two f32 -> one dword of bf16 pair (backend should match v_cvt_pk_bf16_f32)
__device__ __forceinline__ uint32_t pk_bf16(float a, float b) {
  bf16x2 h;
  h[0] = (bf16)a;
  h[1] = (bf16)b;
  return __builtin_bit_cast(uint32_t, h);
}

// ---------------- workspace layout (bytes) ----------------
static const size_t OFF_COUNTS    = 0;                                    // (NEXP+1) ints
static const size_t OFF_SLOT_TOK  = 256;                                  // NEXP*CAP ints
static const size_t OFF_SLOT_PROB = OFF_SLOT_TOK  + (size_t)NEXP*CAP*4;   // NEXP*CAP floats
static const size_t OFF_TOK_SLOT  = OFF_SLOT_PROB + (size_t)NEXP*CAP*4;   // T_ASSIGN ints
static const size_t OFF_ACT       = OFF_TOK_SLOT  + (size_t)T_ASSIGN*4;   // NEXP*CAP*INTER bf16
static const size_t OFF_Y         = OFF_ACT + (size_t)NEXP*CAP*INTER*2;   // NEXP*CAP*DIM f32

#define LDA 40         // LDS row stride in bf16 elements (80B, 16B-aligned, bank-skewed)
#define TILE (128*LDA) // one stage buffer

// ---------------- fragment loads (ISA 7.12.2 16-bit layouts) ----------------
__device__ __forceinline__ v16bf load_frag_a(const bf16* Ls, int rowBase, int lane) {
  const int m  = lane & 15;
  const int kb = (lane >> 4) << 3;           // 0 or 8 elements
  const bf16* p = Ls + (rowBase + m) * LDA;
  FragU f;
  f.q[0] = *(const uint4*)(p + kb);          // K = kb .. kb+7
  f.q[1] = *(const uint4*)(p + 16 + kb);     // K = 16+kb .. 23+kb
  return f.v;
}
__device__ __forceinline__ v16bf load_frag_b(const bf16* Ls, int colBase, int lane) {
  const int n  = lane & 15;
  const int kb = (lane >> 4) << 3;           // 0 or 8 elements
  const bf16* p = Ls + (colBase + n) * LDA + kb;
  FragU f;
  f.q[0] = *(const uint4*)(p);
  f.q[1] = *(const uint4*)(p + 16);
  return f.v;
}

// ---------------- K0: init ----------------
__global__ void moe_init(int* __restrict__ counts, int* __restrict__ slot_tok,
                         float* __restrict__ slot_prob) {
  int i = blockIdx.x * blockDim.x + threadIdx.x;
  if (i < NEXP * CAP) { slot_tok[i] = -1; slot_prob[i] = 0.0f; }
  if (i <= NEXP) counts[i] = 0;
}

// ---------------- K1: routing / slot assignment ----------------
__global__ void moe_route(const unsigned char* __restrict__ mask,
                          const float* __restrict__ wts,
                          const int* __restrict__ indices,
                          int* __restrict__ counts, int* __restrict__ slot_tok,
                          float* __restrict__ slot_prob, int* __restrict__ tok_slot) {
  int i = blockIdx.x * blockDim.x + threadIdx.x;
  if (i >= T_ASSIGN) return;
  const int t = i / TOPK;
  int dst = -1;
  const int e = indices[i];
  if (mask[t] && (unsigned)e < (unsigned)NEXP) {
    int slot = atomicAdd(&counts[e], 1);
    if (slot < CAP) {
      dst = e * CAP + slot;
      slot_tok[dst]  = t;
      slot_prob[dst] = wts[i];
    }
  }
  tok_slot[i] = dst;
}

// ---------------- K2: grouped GEMM1 (x @ gup) + fused quick-GEGLU ----------------
__global__ __launch_bounds__(256)
void moe_gemm1_act(const float* __restrict__ x, const float* __restrict__ gup,
                   const int* __restrict__ counts, const int* __restrict__ slot_tok,
                   const float* __restrict__ slot_prob, bf16* __restrict__ act) {
  const int e = blockIdx.z;
  int cnt = counts[e]; cnt = (cnt > CAP) ? CAP : cnt;
  const int mBlk = blockIdx.y * 128;
  if (mBlk >= cnt) return;                 // block-uniform: EXEC stays full for WMMA
  const int n0   = blockIdx.x * 128;       // column base in h (0..2815)
  const int tid  = threadIdx.x;
  const int lane = tid & 31;
  const int wave = tid >> 5;
  const int wm   = wave & 3;               // 4 wave-rows (32 M each)
  const int wn   = wave >> 2;              // 2 wave-cols (64 N each)

  __shared__ __align__(16) bf16 As[2 * TILE];
  __shared__ __align__(16) bf16 Bs[2 * TILE];

  const int slotBase = e * CAP + mBlk;

  // A-gather mapping: 32 rows/pass x 8 chunks of float4 (4 passes)
  const int arow   = tid >> 3;             // 0..31
  const int achunk = tid & 7;              // 0..7
  int tokR[4];
  float mskR[4];
#pragma unroll
  for (int p = 0; p < 4; ++p) {
    int tk = slot_tok[slotBase + p * 32 + arow];
    mskR[p] = (tk >= 0) ? 1.0f : 0.0f;
    tokR[p] = (tk >= 0) ? tk : 0;          // clamped: loads stay unpredicated
  }

  // B mapping: 4 cols x k-pairs; 2 passes cover 32 k-rows
  const int bcol = (tid & 31) << 2;        // 0..124
  const int bkr  = (tid >> 5) << 1;        // 0,2,..,14

  v8f acc[2][4] = {};

  const size_t gupBase = (size_t)e * DIM * TWO_INTER + n0;

  vf4 aV[4];
  vf4 bV[2][2];

  auto loadTile = [&](int k0) {
#pragma unroll
    for (int p = 0; p < 4; ++p)
      aV[p] = *(const vf4*)(x + (size_t)tokR[p] * DIM + k0 + achunk * 4);
#pragma unroll
    for (int p = 0; p < 2; ++p) {
      const int kr = p * 16 + bkr;
      const float* s0 = gup + gupBase + (size_t)(k0 + kr) * TWO_INTER + bcol;
      bV[p][0] = *(const vf4*)(s0);
      bV[p][1] = *(const vf4*)(s0 + TWO_INTER);
    }
  };
  auto storeTile = [&](int buf) {
    bf16* A_ = As + buf * TILE;
    bf16* B_ = Bs + buf * TILE;
#pragma unroll
    for (int p = 0; p < 4; ++p) {
      const vf4 v = aV[p] * mskR[p];
      vu2 u;
      u[0] = pk_bf16(v.x, v.y);
      u[1] = pk_bf16(v.z, v.w);
      *(vu2*)&A_[(p * 32 + arow) * LDA + achunk * 4] = u;
    }
#pragma unroll
    for (int p = 0; p < 2; ++p) {
      const int kr = p * 16 + bkr;
#pragma unroll
      for (int i = 0; i < 4; ++i)
        *(uint32_t*)&B_[(bcol + i) * LDA + kr] = pk_bf16(bV[p][0][i], bV[p][1][i]);
    }
  };

  loadTile(0);
  storeTile(0);
  int buf = 0;
  for (int k0 = 0; k0 < DIM; k0 += 32) {
    __syncthreads();
    const bool hasNext = (k0 + 32) < DIM;
    if (hasNext) loadTile(k0 + 32);        // issue global loads early (covered by WMMA)

    const bf16* Ac = As + buf * TILE;
    const bf16* Bc = Bs + buf * TILE;
    const v16bf a0 = load_frag_a(Ac, wm * 32 + 0,  lane);
    const v16bf a1 = load_frag_a(Ac, wm * 32 + 16, lane);
#pragma unroll
    for (int nf = 0; nf < 4; ++nf) {
      const v16bf b = load_frag_b(Bc, wn * 64 + nf * 16, lane);
      acc[0][nf] = __builtin_amdgcn_wmma_f32_16x16x32_bf16(false, a0, false, b,
                                                           (short)0, acc[0][nf], false, false);
      acc[1][nf] = __builtin_amdgcn_wmma_f32_16x16x32_bf16(false, a1, false, b,
                                                           (short)0, acc[1][nf], false, false);
    }
    if (hasNext) storeTile(buf ^ 1);       // wait_loadcnt lands here, after WMMA phase
    buf ^= 1;
  }

  // fused quick-GEGLU epilogue. D layout: lane holds n = lane&15, rows r + 8*(lane>>4).
  // gate = even h-column, up = odd -> adjacent lanes; pair via shfl_xor(1).
  const int n      = lane & 15;
  const int mhalf  = (lane >> 4) * 8;
  const bool gateL = ((lane & 1) == 0);
#pragma unroll
  for (int mf = 0; mf < 2; ++mf) {
#pragma unroll
    for (int r = 0; r < 8; ++r) {
      const int rowL = wm * 32 + mf * 16 + mhalf + r;
      const float prob = slot_prob[slotBase + rowL];
#pragma unroll
      for (int nf = 0; nf < 4; ++nf) {
        const float v  = acc[mf][nf][r];
        const float vp = __shfl_xor(v, 1, 32);
        if (gateL) {
          const float gate = fminf(v, LIMIT_C);
          const float up   = fminf(fmaxf(vp, -LIMIT_C), LIMIT_C);
          const float sig  = 1.0f / (1.0f + __expf(-ALPHA_C * gate));
          const float a    = gate * sig * (up + LIN_OFFSET_C) * prob;
          const int hcol   = n0 + wn * 64 + nf * 16 + n;     // even
          act[(size_t)(slotBase + rowL) * INTER + (hcol >> 1)] = (bf16)a;
        }
      }
    }
  }
}

// ---------------- K3: grouped GEMM2 (act @ down) -> y ----------------
__global__ __launch_bounds__(256)
void moe_gemm2(const bf16* __restrict__ act, const float* __restrict__ down,
               const int* __restrict__ counts, float* __restrict__ y) {
  const int e = blockIdx.z;
  int cnt = counts[e]; cnt = (cnt > CAP) ? CAP : cnt;
  const int mBlk = blockIdx.y * 128;
  if (mBlk >= cnt) return;
  const int n0   = blockIdx.x * 128;
  const int tid  = threadIdx.x;
  const int lane = tid & 31;
  const int wave = tid >> 5;
  const int wm   = wave & 3;
  const int wn   = wave >> 2;

  __shared__ __align__(16) bf16 As[2 * TILE];
  __shared__ __align__(16) bf16 Bs[2 * TILE];

  const int slotBase = e * CAP + mBlk;

  // A: bf16 rows, 64 rows/pass x 4 chunks of 8 bf16 (2 passes)
  const int arow   = tid >> 2;             // 0..63
  const int achunk = tid & 3;              // 0..3
  const bf16* actRow[2];
#pragma unroll
  for (int p = 0; p < 2; ++p)
    actRow[p] = act + (size_t)(slotBase + p * 64 + arow) * INTER + achunk * 8;

  const int bcol = (tid & 31) << 2;
  const int bkr  = (tid >> 5) << 1;

  v8f acc[2][4] = {};

  const size_t downBase = (size_t)e * INTER * DIM + n0;

  uint4 aU[2];
  vf4 bV[2][2];

  auto loadTile = [&](int k0) {
#pragma unroll
    for (int p = 0; p < 2; ++p)
      aU[p] = *(const uint4*)(actRow[p] + k0);
#pragma unroll
    for (int p = 0; p < 2; ++p) {
      const int kr = p * 16 + bkr;
      const float* s0 = down + downBase + (size_t)(k0 + kr) * DIM + bcol;
      bV[p][0] = *(const vf4*)(s0);
      bV[p][1] = *(const vf4*)(s0 + DIM);
    }
  };
  auto storeTile = [&](int buf) {
    bf16* A_ = As + buf * TILE;
    bf16* B_ = Bs + buf * TILE;
#pragma unroll
    for (int p = 0; p < 2; ++p)
      *(uint4*)&A_[(p * 64 + arow) * LDA + achunk * 8] = aU[p];
#pragma unroll
    for (int p = 0; p < 2; ++p) {
      const int kr = p * 16 + bkr;
#pragma unroll
      for (int i = 0; i < 4; ++i)
        *(uint32_t*)&B_[(bcol + i) * LDA + kr] = pk_bf16(bV[p][0][i], bV[p][1][i]);
    }
  };

  loadTile(0);
  storeTile(0);
  int buf = 0;
  for (int k0 = 0; k0 < INTER; k0 += 32) {
    __syncthreads();
    const bool hasNext = (k0 + 32) < INTER;
    if (hasNext) loadTile(k0 + 32);

    const bf16* Ac = As + buf * TILE;
    const bf16* Bc = Bs + buf * TILE;
    const v16bf a0 = load_frag_a(Ac, wm * 32 + 0,  lane);
    const v16bf a1 = load_frag_a(Ac, wm * 32 + 16, lane);
#pragma unroll
    for (int nf = 0; nf < 4; ++nf) {
      const v16bf b = load_frag_b(Bc, wn * 64 + nf * 16, lane);
      acc[0][nf] = __builtin_amdgcn_wmma_f32_16x16x32_bf16(false, a0, false, b,
                                                           (short)0, acc[0][nf], false, false);
      acc[1][nf] = __builtin_amdgcn_wmma_f32_16x16x32_bf16(false, a1, false, b,
                                                           (short)0, acc[1][nf], false, false);
    }
    if (hasNext) storeTile(buf ^ 1);
    buf ^= 1;
  }

  // y is write-once/read-once streaming: non-temporal stores keep L2 for weights/act
  const int n     = lane & 15;
  const int mhalf = (lane >> 4) * 8;
#pragma unroll
  for (int mf = 0; mf < 2; ++mf) {
#pragma unroll
    for (int nf = 0; nf < 4; ++nf) {
      const int col = n0 + wn * 64 + nf * 16 + n;
#pragma unroll
      for (int r = 0; r < 8; ++r) {
        const int rowL = wm * 32 + mf * 16 + mhalf + r;
        __builtin_nontemporal_store(acc[mf][nf][r], &y[(size_t)(slotBase + rowL) * DIM + col]);
      }
    }
  }
}

// ---------------- K4: combine (deterministic fixed-order sum over top-k) ----------------
__global__ void moe_combine(const float* __restrict__ y, const int* __restrict__ tok_slot,
                            float* __restrict__ out) {
  const size_t i   = (size_t)blockIdx.x * blockDim.x + threadIdx.x;  // one float4 each
  const size_t idx = i << 2;
  const int t = (int)(idx / DIM);
  const int c = (int)(idx % DIM);
  vf4 s = {0.f, 0.f, 0.f, 0.f};
#pragma unroll
  for (int k = 0; k < TOPK; ++k) {
    const int sl = tok_slot[t * TOPK + k];
    if (sl >= 0) {
      const vf4 v = __builtin_nontemporal_load((const vf4*)(y + (size_t)sl * DIM + c));
      s += v;
    }
  }
  __builtin_nontemporal_store(s, (vf4*)(out + idx));
}

// ---------------- launch ----------------
extern "C" void kernel_launch(void* const* d_in, const int* in_sizes, int n_in,
                              void* d_out, int out_size, void* d_ws, size_t ws_size,
                              hipStream_t stream) {
  (void)in_sizes; (void)n_in; (void)out_size; (void)ws_size;

  const float*         x    = (const float*)d_in[0];
  const unsigned char* mask = (const unsigned char*)d_in[1];
  const float*         wts  = (const float*)d_in[2];
  const int*           idx  = (const int*)d_in[3];
  const float*         gup  = (const float*)d_in[4];
  const float*         down = (const float*)d_in[5];
  float*               out  = (float*)d_out;

  char* ws = (char*)d_ws;
  int*   counts    = (int*)  (ws + OFF_COUNTS);
  int*   slot_tok  = (int*)  (ws + OFF_SLOT_TOK);
  float* slot_prob = (float*)(ws + OFF_SLOT_PROB);
  int*   tok_slot  = (int*)  (ws + OFF_TOK_SLOT);
  bf16*  act       = (bf16*) (ws + OFF_ACT);
  float* y         = (float*)(ws + OFF_Y);

  moe_init<<<(NEXP * CAP + 255) / 256, 256, 0, stream>>>(counts, slot_tok, slot_prob);
  moe_route<<<(T_ASSIGN + 255) / 256, 256, 0, stream>>>(mask, wts, idx, counts,
                                                        slot_tok, slot_prob, tok_slot);

  dim3 g1(TWO_INTER / 128, CAP / 128, NEXP);   // 22 x 8 x 32
  moe_gemm1_act<<<g1, 256, 0, stream>>>(x, gup, counts, slot_tok, slot_prob, act);

  dim3 g2(DIM / 128, CAP / 128, NEXP);         // 16 x 8 x 32
  moe_gemm2<<<g2, 256, 0, stream>>>(act, down, counts, y);

  moe_combine<<<(unsigned)(((size_t)N_TOK * DIM / 4) / 256), 256, 0, stream>>>(y, tok_slot, out);
}
